// WSDDN_76579266888156
// MI455X (gfx1250) — matrix operand order repaired and checked by reading, not compiled, not run on hardware
//
#include <hip/hip_runtime.h>
#include <hip/hip_bf16.h>

// ---------------- problem constants ----------------
#define CCH 512      // channels
#define HX  24
#define HY  24
#define RR  512      // rois
#define KK  20       // classes
#define DD  4096     // hidden dim (= C*2*4); GEMM N and K are this, compile-time
#define NEGV (-1e30f)

typedef __attribute__((ext_vector_type(16))) _Float16 v16h;
typedef __attribute__((ext_vector_type(8)))  _Float16 v8h;
typedef __attribute__((ext_vector_type(8)))  float    v8f;

// ================= 1. ROI adaptive max pool -> f16 (R x 4096) ================
__global__ __launch_bounds__(256)
void k_roi_pool(const float* __restrict__ feat, const int* __restrict__ rois,
                _Float16* __restrict__ Af) {
    const int r   = blockIdx.x;
    const int tid = threadIdx.x;
    const int x = rois[r * 4 + 0];
    const int y = rois[r * 4 + 1];
    const int w = rois[r * 4 + 2];
    const int h = rois[r * 4 + 3];
    for (int c = tid; c < CCH; c += 256) {
        const float* fc = feat + (size_t)c * (HX * HY);
        #pragma unroll
        for (int i = 0; i < 2; ++i) {
            const int sx = x + (i * w) / 2;
            const int ex = x + ((i + 1) * w + 1) / 2;   // ceil
            #pragma unroll
            for (int j = 0; j < 4; ++j) {
                const int sy = y + (j * h) / 4;
                const int ey = y + ((j + 1) * h + 3) / 4;  // ceil
                float m = NEGV;
                for (int px = sx; px < ex; ++px)
                    for (int py = sy; py < ey; ++py)
                        m = fmaxf(m, fc[px * HY + py]);
                Af[(size_t)r * DD + c * 8 + i * 4 + j] = (_Float16)m;
            }
        }
    }
}

// ================= 2. WMMA GEMM: C(512 x 4096) = A(f16) @ B(fp32) + bias =====
// block tile 128x128, BK=32, 256 threads (8 waves), wave tile 32x64.
// Register-prefetch + LDS double-buffer pipeline, one barrier per K-step.
#define LDT 40   // padded LDS row length in halves (80B rows keep 16B alignment)
__global__ __launch_bounds__(256)
void k_gemm_wmma(const _Float16* __restrict__ A, const float* __restrict__ B,
                 const float* __restrict__ bias,
                 _Float16* __restrict__ out16, float* __restrict__ out32) {
    __shared__ alignas(16) _Float16 As[2][128 * LDT];
    __shared__ alignas(16) _Float16 Bs[2][128 * LDT];

    const int tid  = threadIdx.x;
    const int bm   = blockIdx.y * 128;
    const int bn   = blockIdx.x * 128;
    const int wave = tid >> 5;
    const int lane = tid & 31;
    const int wm   = (wave & 3) * 32;   // wave M offset in block tile
    const int wn   = (wave >> 2) * 64;  // wave N offset
    const int lrow = lane >> 4;         // 0/1 half-wave
    const int lcol = lane & 15;

    v8f acc[2][4];
    const v8f vzero = {0.f, 0.f, 0.f, 0.f, 0.f, 0.f, 0.f, 0.f};
    #pragma unroll
    for (int mt = 0; mt < 2; ++mt)
        #pragma unroll
        for (int nt = 0; nt < 4; ++nt) acc[mt][nt] = vzero;

    const int am = tid >> 1;          // A row 0..127
    const int ah = tid & 1;           // which 16-half chunk of the 32-wide slab
    const int bc = tid & 127;         // B col 0..127
    const int br = (tid >> 7) * 16;   // B row base 0/16

    // per-thread base pointers (k0 added inside; row offsets are immediates)
    const _Float16* aptr = A + (size_t)(bm + am) * DD + ah * 16;
    const float*    bptr = B + (size_t)br * DD + bn + bc;

    // staging registers
    uint4 ar0, ar1;
    float br_f[16];

    auto load_tile = [&](int k0) {
        const uint4* as = (const uint4*)(aptr + k0);
        ar0 = as[0];
        ar1 = as[1];
        const float* bs = bptr + (size_t)k0 * DD;
        #pragma unroll
        for (int i = 0; i < 16; ++i) br_f[i] = bs[(size_t)i * DD];
    };
    auto store_tile = [&](int buf) {
        _Float16* ad = As[buf] + am * LDT + ah * 16;
        *(uint4*)(ad)     = ar0;
        *(uint4*)(ad + 8) = ar1;
        _Float16* bd = Bs[buf] + bc * LDT + br;
        #pragma unroll
        for (int i = 0; i < 8; ++i) {
            union { _Float16 h[2]; unsigned int u; } pk;
            pk.h[0] = (_Float16)br_f[2 * i];
            pk.h[1] = (_Float16)br_f[2 * i + 1];
            *(unsigned int*)(bd + 2 * i) = pk.u;
        }
    };

    // ---- pipeline prologue ----
    load_tile(0);
    store_tile(0);
    load_tile(32);
    __syncthreads();

    int cur = 0;
    for (int k0 = 0; k0 < DD; k0 += 32) {
        if (k0 + 32 < DD) {
            store_tile(cur ^ 1);              // stage tile k0+32 into other buffer
            if (k0 + 64 < DD) {
                load_tile(k0 + 64);           // fetch tile k0+64 into registers
                __builtin_prefetch(bptr + (size_t)(k0 + 96) * DD, 0, 1);
            }
        }
        // --- fragments (ISA 7.12.2 layouts) from current buffer ---
        v16h afrag[2], bfrag[4];
        #pragma unroll
        for (int mt = 0; mt < 2; ++mt) {
            const _Float16* p = As[cur] + (wm + mt * 16 + lcol) * LDT + lrow * 8;
            v8h lo = *(const v8h*)(p);
            v8h hi = *(const v8h*)(p + 16);
            #pragma unroll
            for (int e = 0; e < 8; ++e) { afrag[mt][e] = lo[e]; afrag[mt][8 + e] = hi[e]; }
        }
        #pragma unroll
        for (int nt = 0; nt < 4; ++nt) {
            const _Float16* p = Bs[cur] + (wn + nt * 16 + lcol) * LDT + lrow * 8;
            v8h lo = *(const v8h*)(p);
            v8h hi = *(const v8h*)(p + 16);
            #pragma unroll
            for (int e = 0; e < 8; ++e) { bfrag[nt][e] = lo[e]; bfrag[nt][8 + e] = hi[e]; }
        }
        #pragma unroll
        for (int mt = 0; mt < 2; ++mt)
            #pragma unroll
            for (int nt = 0; nt < 4; ++nt)
                acc[mt][nt] = __builtin_amdgcn_wmma_f32_16x16x32_f16(
                    false, afrag[mt], false, bfrag[nt], (short)0, acc[mt][nt],
                    false, false);
        __syncthreads();
        cur ^= 1;
    }

    // --- epilogue: +bias, store f16 and/or f32 ---
    #pragma unroll
    for (int mt = 0; mt < 2; ++mt) {
        #pragma unroll
        for (int nt = 0; nt < 4; ++nt) {
            const int gm0 = bm + wm + mt * 16 + lrow * 8;  // rows gm0..gm0+7
            const int gn  = bn + wn + nt * 16 + lcol;
            const float bv = bias[gn];
            #pragma unroll
            for (int v = 0; v < 8; ++v) {
                const float val = acc[mt][nt][v] + bv;
                const size_t off = (size_t)(gm0 + v) * DD + gn;
                if (out16) out16[off] = (_Float16)val;
                if (out32) out32[off] = val;
            }
        }
    }
}

// ================= 3. fc8 heads: logits (R x 20) for both c and d ============
__global__ __launch_bounds__(32)
void k_logits(const float* __restrict__ fc7,
              const float* __restrict__ W8c, const float* __restrict__ b8c,
              const float* __restrict__ W8d, const float* __restrict__ b8d,
              float* __restrict__ Zc, float* __restrict__ Zd) {
    const int r = blockIdx.x;
    const int k = threadIdx.x;
    if (k >= KK) return;
    const float* f = fc7 + (size_t)r * DD;
    float ac = b8c[k], ad = b8d[k];
    #pragma unroll 8
    for (int d = 0; d < DD; ++d) {
        const float fv = f[d];
        ac += fv * W8c[(size_t)d * KK + k];
        ad += fv * W8d[(size_t)d * KK + k];
    }
    Zc[r * KK + k] = ac;
    Zd[r * KK + k] = ad;
}

// ================= 4. softmax over classes (axis=1), per roi =================
__global__ __launch_bounds__(256)
void k_softmax_rows(const float* __restrict__ Zc, float* __restrict__ Pc) {
    const int r = blockIdx.x * 256 + threadIdx.x;
    if (r >= RR) return;
    float m = NEGV;
    #pragma unroll
    for (int k = 0; k < KK; ++k) m = fmaxf(m, Zc[r * KK + k]);
    float s = 0.f;
    float e[KK];
    #pragma unroll
    for (int k = 0; k < KK; ++k) { e[k] = expf(Zc[r * KK + k] - m); s += e[k]; }
    const float inv = 1.f / s;
    #pragma unroll
    for (int k = 0; k < KK; ++k) Pc[r * KK + k] = e[k] * inv;
}

// ====== 5. per-class: softmax over rois (axis=0) + scores + colsum + argmin ==
__global__ __launch_bounds__(256)
void k_softmax_cols(const float* __restrict__ Zd, const float* __restrict__ Pc,
                    float* __restrict__ scores, float* __restrict__ colsum,
                    int* __restrict__ hi) {
    const int k   = blockIdx.x;
    const int tid = threadIdx.x;
    __shared__ float red[256];
    __shared__ int   redi[256];

    const float v0 = Zd[tid * KK + k];
    const float v1 = Zd[(tid + 256) * KK + k];
    // max
    red[tid] = fmaxf(v0, v1);
    __syncthreads();
    for (int s = 128; s > 0; s >>= 1) {
        if (tid < s) red[tid] = fmaxf(red[tid], red[tid + s]);
        __syncthreads();
    }
    const float m = red[0];
    __syncthreads();
    // sum
    const float e0 = expf(v0 - m), e1 = expf(v1 - m);
    red[tid] = e0 + e1;
    __syncthreads();
    for (int s = 128; s > 0; s >>= 1) {
        if (tid < s) red[tid] += red[tid + s];
        __syncthreads();
    }
    const float inv = 1.f / red[0];
    __syncthreads();
    // scores = fc8c * fc8d ; column sum
    const float p0 = (e0 * inv) * Pc[tid * KK + k];
    const float p1 = (e1 * inv) * Pc[(tid + 256) * KK + k];
    scores[tid * KK + k]         = p0;
    scores[(tid + 256) * KK + k] = p1;
    red[tid] = p0 + p1;
    __syncthreads();
    for (int s = 128; s > 0; s >>= 1) {
        if (tid < s) red[tid] += red[tid + s];
        __syncthreads();
    }
    if (tid == 0) colsum[k] = red[0];
    __syncthreads();
    // stable argmin (jnp.argsort ascending -> idx[0] = min, ties -> lower idx)
    float bv; int bi;
    if (p0 <= p1) { bv = p0; bi = tid; } else { bv = p1; bi = tid + 256; }
    red[tid] = bv; redi[tid] = bi;
    __syncthreads();
    for (int s = 128; s > 0; s >>= 1) {
        if (tid < s) {
            const float ov = red[tid + s];
            const int   oi = redi[tid + s];
            if (ov < red[tid] || (ov == red[tid] && oi < redi[tid])) {
                red[tid] = ov; redi[tid] = oi;
            }
        }
        __syncthreads();
    }
    if (tid == 0) hi[k] = redi[0];
}

// ================= 6. spatial regulariser, one block per class ===============
__global__ __launch_bounds__(256)
void k_reg(const float* __restrict__ scores, const float* __restrict__ fc7,
           const int* __restrict__ rois, const float* __restrict__ label,
           const int* __restrict__ hi, float* __restrict__ regk) {
    const int k   = blockIdx.x;
    const int tid = threadIdx.x;
    __shared__ float red[256];

    const int h = hi[k];
    const float hb0 = (float)rois[h * 4 + 0];
    const float hb1 = (float)rois[h * 4 + 1];
    const float hb2 = (float)rois[h * 4 + 2];
    const float hb3 = (float)rois[h * 4 + 3];
    const float a2  = (hb2 - hb0 + 1.f) * (hb3 - hb1 + 1.f);

    // this thread's slice of fc7[hi]
    float hrow[16];
    float hsq = 0.f;
    const float* fh = fc7 + (size_t)h * DD;
    #pragma unroll
    for (int t = 0; t < 16; ++t) {
        hrow[t] = fh[tid + t * 256];
        hsq += hrow[t] * hrow[t];
    }

    float acc = 0.f;
    for (int r = 0; r < RR; ++r) {
        if (r == h) continue;
        const float r0 = (float)rois[r * 4 + 0];
        const float r1 = (float)rois[r * 4 + 1];
        const float r2 = (float)rois[r * 4 + 2];
        const float r3 = (float)rois[r * 4 + 3];
        const float ix1 = fmaxf(r0, hb0), iy1 = fmaxf(r1, hb1);
        const float ix2 = fminf(r2, hb2), iy2 = fminf(r3, hb3);
        const float inter = fmaxf(ix2 - ix1 + 1.f, 0.f) * fmaxf(iy2 - iy1 + 1.f, 0.f);
        const float a1 = (r2 - r0 + 1.f) * (r3 - r1 + 1.f);
        const float iou = inter / (a1 + a2 - inter);
        const float sr = scores[r * KK + k];
        const float w  = sr * sr;
        if (iou > 0.6f) {  // uniform across block
            const float* fr = fc7 + (size_t)r * DD;
            float d2 = 0.f;
            #pragma unroll
            for (int t = 0; t < 16; ++t) {
                const float v = fr[tid + t * 256] - hrow[t];
                d2 += v * v;
            }
            acc += w * d2;
        } else {
            acc += w * hsq;  // (0*fc7_r - fc7_hi)^2 contribution
        }
    }
    red[tid] = acc;
    __syncthreads();
    for (int s = 128; s > 0; s >>= 1) {
        if (tid < s) red[tid] += red[tid + s];
        __syncthreads();
    }
    if (tid == 0) regk[k] = 0.5f * label[k] * red[0];
}

// ================= 7. finalize: BCE + sum(reg) ===============================
__global__ __launch_bounds__(32)
void k_final(const float* __restrict__ colsum, const float* __restrict__ label,
             const float* __restrict__ regk, float* __restrict__ out) {
    const int k = threadIdx.x;
    if (k >= KK) return;
    float o = colsum[k];
    o = fminf(fmaxf(o, 1e-12f), 1.f - 1e-12f);
    const float lb = label[k];
    const float bce = -(lb * logf(o) + (1.f - lb) * logf(1.f - o));
    float reg = 0.f;
    for (int j = 0; j < KK; ++j) reg += regk[j];
    out[k] = bce + reg;
}

// ================= launcher ==================================================
extern "C" void kernel_launch(void* const* d_in, const int* in_sizes, int n_in,
                              void* d_out, int out_size, void* d_ws, size_t ws_size,
                              hipStream_t stream) {
    const float* features = (const float*)d_in[0];
    const int*   rois     = (const int*)d_in[1];
    const float* label    = (const float*)d_in[2];
    const float* W6       = (const float*)d_in[3];
    const float* b6       = (const float*)d_in[4];
    const float* W7       = (const float*)d_in[5];
    const float* b7       = (const float*)d_in[6];
    const float* W8c      = (const float*)d_in[7];
    const float* b8c      = (const float*)d_in[8];
    const float* W8d      = (const float*)d_in[9];
    const float* b8d      = (const float*)d_in[10];
    float* out = (float*)d_out;

    // workspace carve-up
    char* p = (char*)d_ws;
    _Float16* Af   = (_Float16*)p;  p += (size_t)RR * DD * sizeof(_Float16);
    _Float16* fc6h = (_Float16*)p;  p += (size_t)RR * DD * sizeof(_Float16);
    float* fc7     = (float*)p;     p += (size_t)RR * DD * sizeof(float);
    float* Zc      = (float*)p;     p += (size_t)RR * KK * sizeof(float);
    float* Zd      = (float*)p;     p += (size_t)RR * KK * sizeof(float);
    float* Pc      = (float*)p;     p += (size_t)RR * KK * sizeof(float);
    float* scores  = (float*)p;     p += (size_t)RR * KK * sizeof(float);
    float* colsum  = (float*)p;     p += 32 * sizeof(float);
    int*   hi      = (int*)p;       p += 32 * sizeof(int);
    float* regk    = (float*)p;     p += 32 * sizeof(float);

    // 1. ROI max-pool -> f16 activations
    k_roi_pool<<<RR, 256, 0, stream>>>(features, rois, Af);

    // 2. fc6 = Af @ W6 + b6  (store f16 for next GEMM)
    k_gemm_wmma<<<dim3(DD / 128, RR / 128), 256, 0, stream>>>(
        Af, W6, b6, fc6h, nullptr);

    // 3. fc7 = fc6 @ W7 + b7 (store fp32 for heads + regulariser)
    k_gemm_wmma<<<dim3(DD / 128, RR / 128), 256, 0, stream>>>(
        fc6h, W7, b7, nullptr, fc7);

    // 4. head logits
    k_logits<<<RR, 32, 0, stream>>>(fc7, W8c, b8c, W8d, b8d, Zc, Zd);

    // 5. softmax over classes
    k_softmax_rows<<<RR / 256, 256, 0, stream>>>(Zc, Pc);

    // 6. softmax over rois + scores + colsum + argmin
    k_softmax_cols<<<KK, 256, 0, stream>>>(Zd, Pc, scores, colsum, hi);

    // 7. spatial regulariser per class
    k_reg<<<KK, 256, 0, stream>>>(scores, fc7, rois, label, hi, regk);

    // 8. BCE + reg
    k_final<<<1, 32, 0, stream>>>(colsum, label, regk, out);
}
